// AttributeDecoder_43130061586963
// MI455X (gfx1250) — compile-verified, matrix operand order checked by reading, not run on hardware
//
#include <hip/hip_runtime.h>

// ---------------------------------------------------------------------------
// 2-layer GCN for MI455X (gfx1250, wave32).
//   layer1: h   = relu( D^-1/2 (A+I) D^-1/2 (z @ W1) + b1 )
//   layer2: out =       D^-1/2 (A+I) D^-1/2 (h) @ W2 + b2      (agg-then-GEMM)
// GEMMs: v_wmma_f32_16x16x32_bf16 with Ootomo-style hi/lo bf16 split of fp32
// (3 WMMAs per tile-K-step => ~fp32 accuracy at bf16 WMMA rates).
// W is pre-packed once into the exact B-fragment dword order (hi & lo), so the
// GEMM hot loop is: 4x b128 A loads + per N-tile 4x b128 B loads + 3 WMMAs.
// ---------------------------------------------------------------------------

typedef __attribute__((ext_vector_type(16))) __bf16   v16bf;
typedef __attribute__((ext_vector_type(8)))  float    v8f;
typedef __attribute__((ext_vector_type(8)))  unsigned v8u;

constexpr int IN_C  = 512;
constexpr int HID   = 128;
constexpr int OUT_C = 512;

// ---------------- degree / normalization ----------------------------------

__global__ __launch_bounds__(256) void deg_init_k(float* __restrict__ deg, int n) {
    int i = blockIdx.x * blockDim.x + threadIdx.x;
    if (i < n) deg[i] = 1.0f;                       // self loop
}

__global__ __launch_bounds__(256) void deg_count_k(const long long* __restrict__ dst,
                                                   float* __restrict__ deg,
                                                   long long nE) {
    long long e = (long long)blockIdx.x * blockDim.x + threadIdx.x;
    if (e < nE) unsafeAtomicAdd(&deg[dst[e]], 1.0f);
}

__global__ __launch_bounds__(256) void deg_rsqrt_k(float* __restrict__ deg, int n) {
    int i = blockIdx.x * blockDim.x + threadIdx.x;
    if (i < n) deg[i] = rsqrtf(deg[i]);             // deg >= 1 always
}

// ---------------- W pre-pack into B-fragment layout ------------------------
// B fragment (16-bit, 32x16, mirroring documented A layout with N per lane):
// lane = column n (0..15), half h = lane>>4 selects K-offset 0/8, VGPR v holds
// K pair Kidx = kt*32 + (v>=4?16:0) + 8h + 2(v&3).
// Packed dword stream: dst[(((kt*NCOLS + n)*2 + h)*8 + v]  -> a lane reads its
// whole fragment as 8 contiguous dwords (two b128 loads).

template <int K, int NCOLS>
__global__ __launch_bounds__(256) void pack_B_k(const float* __restrict__ W,
                                                unsigned* __restrict__ hi,
                                                unsigned* __restrict__ lo) {
    const int total = (K / 32) * NCOLS * 16;
    int idx = blockIdx.x * blockDim.x + threadIdx.x;
    if (idx >= total) return;
    const int v    = idx & 7;
    const int h    = (idx >> 3) & 1;
    const int rest = idx >> 4;              // kt*NCOLS + n
    const int n    = rest % NCOLS;
    const int kt   = rest / NCOLS;
    const int Kidx = kt * 32 + ((v & 4) << 2) + (h << 3) + ((v & 3) << 1);

    const float w0 = W[(size_t)Kidx * NCOLS + n];
    const float w1 = W[(size_t)(Kidx + 1) * NCOLS + n];
    const __bf16 h0 = (__bf16)w0, h1 = (__bf16)w1;
    const __bf16 l0 = (__bf16)(w0 - (float)h0), l1 = (__bf16)(w1 - (float)h1);
    union { __bf16 b[2]; unsigned u; } ph, pl;
    ph.b[0] = h0; ph.b[1] = h1;
    pl.b[0] = l0; pl.b[1] = l1;
    hi[idx] = ph.u;                         // dst offset == idx by construction
    lo[idx] = pl.u;
}

// ---------------- WMMA GEMM: C[M x NCOLS] = X[M x K] @ W + (bias) ----------
// One wave computes a 16 x 128 strip (8 N-tiles): A fragment built once per
// K-step (4x b128 + hi/lo split), reused across 8 pre-packed B fragments.

__device__ inline void cvt2(float x, float y, v16bf& hi, v16bf& lo, int e) {
    const __bf16 hx = (__bf16)x, hy = (__bf16)y;
    hi[e]     = hx;
    hi[e + 1] = hy;
    lo[e]     = (__bf16)(x - (float)hx);
    lo[e + 1] = (__bf16)(y - (float)hy);
}

template <int K, int NCOLS>
__global__ __launch_bounds__(256) void gemm_packed_k(const float* __restrict__ X,
                                                     const unsigned* __restrict__ Bhi,
                                                     const unsigned* __restrict__ Blo,
                                                     const float* __restrict__ bias,
                                                     float* __restrict__ C, int M) {
    constexpr int NT = 8;                   // N-tiles per wave (128 cols)
    constexpr int CG = NCOLS / (16 * NT);   // column groups
    const int lane   = threadIdx.x & 31;
    const int wave   = threadIdx.x >> 5;
    const int Mtiles = (M + 15) >> 4;
    const int strip  = blockIdx.x * 8 + wave;
    const int tileM  = strip / CG;
    const int cg     = strip % CG;
    if (tileM >= Mtiles) return;

    const int n0     = cg * (16 * NT);
    const int mRow   = tileM * 16 + (lane & 15);
    const int mLoad  = mRow < M ? mRow : (M - 1);
    const int kHalf  = (lane >> 4) << 3;    // 0 or 8
    const int nLane  = lane & 15;

    v8f acc[NT];
#pragma unroll
    for (int t = 0; t < NT; ++t) acc[t] = (v8f){};

    const float* arow = X + (size_t)mLoad * K;

    for (int kt = 0; kt < K / 32; ++kt) {
        const int k0 = kt * 32;
        v16bf a_hi, a_lo;
        {
            const float4 q0 = *(const float4*)(arow + k0 + kHalf);
            const float4 q1 = *(const float4*)(arow + k0 + kHalf + 4);
            const float4 q2 = *(const float4*)(arow + k0 + 16 + kHalf);
            const float4 q3 = *(const float4*)(arow + k0 + 16 + kHalf + 4);
            cvt2(q0.x, q0.y, a_hi, a_lo, 0);
            cvt2(q0.z, q0.w, a_hi, a_lo, 2);
            cvt2(q1.x, q1.y, a_hi, a_lo, 4);
            cvt2(q1.z, q1.w, a_hi, a_lo, 6);
            cvt2(q2.x, q2.y, a_hi, a_lo, 8);
            cvt2(q2.z, q2.w, a_hi, a_lo, 10);
            cvt2(q3.x, q3.y, a_hi, a_lo, 12);
            cvt2(q3.z, q3.w, a_hi, a_lo, 14);
        }
#pragma unroll
        for (int t = 0; t < NT; ++t) {
            const int n = n0 + t * 16 + nLane;
            const size_t off = (((size_t)kt * NCOLS + n) * 2 + (lane >> 4)) * 8;
            const v16bf b_hi = __builtin_bit_cast(v16bf, *(const v8u*)(Bhi + off));
            const v16bf b_lo = __builtin_bit_cast(v16bf, *(const v8u*)(Blo + off));
            acc[t] = __builtin_amdgcn_wmma_f32_16x16x32_bf16(false, a_hi, false, b_hi,
                                                             (short)0, acc[t], false, false);
            acc[t] = __builtin_amdgcn_wmma_f32_16x16x32_bf16(false, a_hi, false, b_lo,
                                                             (short)0, acc[t], false, false);
            acc[t] = __builtin_amdgcn_wmma_f32_16x16x32_bf16(false, a_lo, false, b_hi,
                                                             (short)0, acc[t], false, false);
        }
    }

    // store: C/D layout -> row = tileM*16 + 8*(lane>>4) + r, col = n
    const int  rBase = tileM * 16 + ((lane >> 4) << 3);
    const bool full  = (tileM * 16 + 16) <= M;    // always true for M=100000
#pragma unroll
    for (int t = 0; t < NT; ++t) {
        const int   n  = n0 + t * 16 + nLane;
        const float bv = bias ? bias[n] : 0.0f;
        float* cp = C + (size_t)rBase * NCOLS + n;
        if (full) {
#pragma unroll
            for (int r = 0; r < 8; ++r) cp[(size_t)r * NCOLS] = acc[t][r] + bv;
        } else {
#pragma unroll
            for (int r = 0; r < 8; ++r)
                if (rBase + r < M) cp[(size_t)r * NCOLS] = acc[t][r] + bv;
        }
    }
}

// ---------------- aggregation ----------------------------------------------

// agg[i,:] = h[i,:] * dinv[i]^2   (self-loop term, also initializes agg)
__global__ __launch_bounds__(256) void selfloop_init_k(const float* __restrict__ h,
                                                       const float* __restrict__ dinv,
                                                       float* __restrict__ agg, int n) {
    int i = blockIdx.x * blockDim.x + threadIdx.x;     // over n * HID/4 float4s
    if (i >= n * (HID / 4)) return;
    const int node = i / (HID / 4);
    float s = dinv[node];
    s = s * s;
    const float4 v = ((const float4*)h)[i];
    float4 o;
    o.x = v.x * s; o.y = v.y * s; o.z = v.z * s; o.w = v.w * s;
    ((float4*)agg)[i] = o;
}

// one wave32 per edge: lanes cover 128 features as float4; L2-resident atomics
__global__ __launch_bounds__(256) void edge_scatter_k(const long long* __restrict__ src,
                                                      const long long* __restrict__ dst,
                                                      const float* __restrict__ dinv,
                                                      const float* __restrict__ h,
                                                      float* __restrict__ agg,
                                                      long long nE) {
    const int lane = threadIdx.x & 31;
    long long w = (long long)blockIdx.x * 8 + (threadIdx.x >> 5);
    const long long stride = (long long)gridDim.x * 8;
    for (long long e = w; e < nE; e += stride) {
        const long long s = src[e];
        const long long d = dst[e];
        const float norm = dinv[s] * dinv[d];
        const float4 v = *(const float4*)(h + (size_t)s * HID + lane * 4);
        float* o = agg + (size_t)d * HID + lane * 4;
        unsafeAtomicAdd(o + 0, v.x * norm);
        unsafeAtomicAdd(o + 1, v.y * norm);
        unsafeAtomicAdd(o + 2, v.z * norm);
        unsafeAtomicAdd(o + 3, v.w * norm);
    }
}

// h[i] = relu(agg[i] + b[f])
__global__ __launch_bounds__(256) void bias_relu_k(float* __restrict__ a,
                                                   const float* __restrict__ b,
                                                   int total) {
    int i = blockIdx.x * blockDim.x + threadIdx.x;
    if (i >= total) return;
    float v = a[i] + b[i & (HID - 1)];
    a[i] = v > 0.0f ? v : 0.0f;
}

// ---------------------------------------------------------------------------

extern "C" void kernel_launch(void* const* d_in, const int* in_sizes, int n_in,
                              void* d_out, int out_size, void* d_ws, size_t ws_size,
                              hipStream_t stream) {
    const float*     z  = (const float*)d_in[0];
    const long long* ei = (const long long*)d_in[1];   // int64 edge_index [2,E]
    const float*     W1 = (const float*)d_in[2];
    const float*     b1 = (const float*)d_in[3];
    const float*     W2 = (const float*)d_in[4];
    const float*     b2 = (const float*)d_in[5];
    float*           out = (float*)d_out;

    const int       nN = in_sizes[0] / IN_C;           // 100000
    const long long nE = (long long)(in_sizes[1] / 2); // 3200000
    const long long* src = ei;
    const long long* dst = ei + nE;

    // workspace: dinv[N] | buf1[N*HID] | buf2[N*HID] | packed W1/W2 (hi+lo)
    float* dinv = (float*)d_ws;
    float* buf1 = dinv + (((size_t)nN + 255) & ~(size_t)255);
    float* buf2 = buf1 + (size_t)nN * HID;
    unsigned* pk = (unsigned*)(buf2 + (size_t)nN * HID);
    const int pkW1 = (IN_C / 32) * HID * 16;           // 32768 dwords
    const int pkW2 = (HID / 32) * OUT_C * 16;          // 32768 dwords
    unsigned* B1hi = pk;
    unsigned* B1lo = B1hi + pkW1;
    unsigned* B2hi = B1lo + pkW1;
    unsigned* B2lo = B2hi + pkW2;

    const int tN   = (nN + 255) / 256;
    const int tNF4 = (nN * (HID / 4) + 255) / 256;
    const int tNF  = (nN * HID + 255) / 256;

    // --- symmetric normalization coefficients -----------------------------
    deg_init_k<<<tN, 256, 0, stream>>>(dinv, nN);
    deg_count_k<<<(int)((nE + 255) / 256), 256, 0, stream>>>(dst, dinv, nE);
    deg_rsqrt_k<<<tN, 256, 0, stream>>>(dinv, nN);

    // --- pre-pack weights into WMMA B-fragment hi/lo bf16 streams ---------
    pack_B_k<IN_C, HID><<<(pkW1 + 255) / 256, 256, 0, stream>>>(W1, B1hi, B1lo);
    pack_B_k<HID, OUT_C><<<(pkW2 + 255) / 256, 256, 0, stream>>>(W2, B2hi, B2lo);

    // --- layer 1: GEMM (512->128) then aggregate at width 128 -------------
    {
        const int strips = ((nN + 15) / 16) * 1;       // CG = 128/128 = 1
        gemm_packed_k<IN_C, HID><<<(strips + 7) / 8, 256, 0, stream>>>(
            z, B1hi, B1lo, nullptr, buf1, nN);
    }
    selfloop_init_k<<<tNF4, 256, 0, stream>>>(buf1, dinv, buf2, nN);
    edge_scatter_k<<<4096, 256, 0, stream>>>(src, dst, dinv, buf1, buf2, nE);
    bias_relu_k<<<tNF, 256, 0, stream>>>(buf2, b1, nN * HID);

    // --- layer 2: aggregate at width 128 FIRST, then GEMM (128->512) ------
    //     valid since A_hat(h W2) == (A_hat h) W2 ; 4x less scatter traffic
    selfloop_init_k<<<tNF4, 256, 0, stream>>>(buf2, dinv, buf1, nN);
    edge_scatter_k<<<4096, 256, 0, stream>>>(src, dst, dinv, buf2, buf1, nE);
    {
        const int strips = ((nN + 15) / 16) * 4;       // CG = 512/128 = 4
        gemm_packed_k<HID, OUT_C><<<(strips + 7) / 8, 256, 0, stream>>>(
            buf1, B2hi, B2lo, b2, out, nN);
    }
}
